// MemoryModule_12489764896963
// MI455X (gfx1250) — matrix operand order repaired and checked by reading, not compiled, not run on hardware
//
#include <hip/hip_runtime.h>
#include <math.h>

// MemoryModule on MI455X (gfx1250, wave32, WMMA).
// B=256, N=512, M=64, C=128. One workgroup per n; per-n working set staged in
// LDS as bf16 (CDNA5: 320KB/WGP), GEMMs via v_wmma_f32_16x16x32_bf16 with f32
// accumulate. A fragments are contiguous 16B ds_load_b128s (B: strided 2B loads) from padded
// LDS buffers (transposed operands staged once at load time).

#define Bsz 256
#define Nsz 512
#define Msz 64
#define Csz 128
#define SHRINK 0.0025f
#define EPSV   1e-12f

typedef __attribute__((ext_vector_type(16))) __bf16 v16bf;
typedef __attribute__((ext_vector_type(8)))  __bf16 v8bf;   // 16 bytes
typedef __attribute__((ext_vector_type(8)))  float  v8f;

extern __shared__ char smemraw[];

// ---- WMMA fragment helpers (layouts per CDNA5 ISA §7.12.2) ----
// A fragment (16x32 tile, row-major bf16 src, leading dim lda, lda%8==0):
// lane L: M = m0+(L&15), half = L>>4; elements are two contiguous 8-elem runs
// at k0+8*half and k0+16+8*half  -> two ds_load_b128.
__device__ __forceinline__ v16bf frag_a(const __bf16* src, int m0, int k0,
                                        int lda, int lane) {
  const __bf16* p = src + (m0 + (lane & 15)) * lda + k0 + ((lane >> 4) << 3);
  v8bf lo = *(const v8bf*)p;
  v8bf hi = *(const v8bf*)(p + 16);
  return __builtin_shufflevector(lo, hi, 0, 1, 2, 3, 4, 5, 6, 7,
                                 8, 9, 10, 11, 12, 13, 14, 15);
}

// B fragment (32x16 tile): B[k][n] = src[k*ldb + n], lane L: N = n0+(L&15),
// half = L>>4; elements i: K = k0+8*half+i (i<8), k0+16+8*half+(i-8) (i>=8).
__device__ __forceinline__ v16bf frag_b(const __bf16* src, int k0, int n0,
                                        int ldb, int lane) {
  const __bf16* p = src + (k0 + ((lane >> 4) << 3)) * ldb + n0 + (lane & 15);
  v16bf b;
#pragma unroll
  for (int i = 0; i < 8; ++i) {
    b[i]     = p[i * ldb];
    b[8 + i] = p[(16 + i) * ldb];
  }
  return b;
}

// D/C fragment store: VGPR i -> M = m0 + 8*(lane>>4) + i, N = n0 + (lane&15).
__device__ __forceinline__ void store_d(float* dst, v8f d, int m0, int n0,
                                        int ld, int lane) {
  const int n  = n0 + (lane & 15);
  const int mb = m0 + ((lane >> 4) << 3);
#pragma unroll
  for (int i = 0; i < 8; ++i) dst[(mb + i) * ld + n] = d[i];
}

__device__ __forceinline__ v8f wmma_bf16(v16bf a, v16bf b, v8f c) {
  return __builtin_amdgcn_wmma_f32_16x16x32_bf16(
      false, a, false, b, (short)0, c, false, false);
}

__device__ __forceinline__ float wave_max(float v) {
#pragma unroll
  for (int off = 16; off > 0; off >>= 1) v = fmaxf(v, __shfl_xor(v, off, 32));
  return v;
}
__device__ __forceinline__ float wave_sum(float v) {
#pragma unroll
  for (int off = 16; off > 0; off >>= 1) v += __shfl_xor(v, off, 32);
  return v;
}

// -------- Phase A LDS layout (bytes; all 16B aligned) --------
#define QB_LD 136   // q   bf16 [256][136]
#define QT_LD 264   // q^T bf16 [128][264]
#define MB_LD 136   // mem bf16 [64][136]
#define SC_LD 256   // scores f32 [64][256]
#define AT_LD 264   // attn bf16 [64][264]
#define AMB_LD 136  // add_mem bf16 [64][136]
#define UW_LD 136   // Uw^T/Ww^T bf16 [128][136]
#define A_QB  0
#define A_QT  69632
#define A_MB  137216
#define A_SC  154624
#define A_AT  220160
#define A_AMF 253952
#define A_AMB 286720
#define A_TOTAL 304128
#define A_UWT A_QT   // recycled after scores stage
#define A_WWT A_SC   // recycled after softmax stage

// ---------------- Phase A: memory update (per n) ----------------
__global__ void mm_update_kernel(const float* __restrict__ Q,
                                 const float* __restrict__ Mem,
                                 const float* __restrict__ Uw,
                                 const float* __restrict__ Ub,
                                 const float* __restrict__ Ww,
                                 const float* __restrict__ Wb,
                                 float* __restrict__ MemNew) {
  const int n    = blockIdx.x;
  const int tid  = threadIdx.x;
  const int lane = tid & 31;
  const int w    = tid >> 5;

  __bf16* qb  = (__bf16*)(smemraw + A_QB);
  __bf16* qT  = (__bf16*)(smemraw + A_QT);
  __bf16* mb  = (__bf16*)(smemraw + A_MB);
  float*  sc  = (float*)(smemraw + A_SC);
  __bf16* at  = (__bf16*)(smemraw + A_AT);
  float*  amf = (float*)(smemraw + A_AMF);
  __bf16* amb = (__bf16*)(smemraw + A_AMB);
  __bf16* uwT = (__bf16*)(smemraw + A_UWT);
  __bf16* wwT = (__bf16*)(smemraw + A_WWT);

  // warm L2 with the (shared) gate weights
#pragma unroll
  for (int p = 0; p < 4; ++p) {
    __builtin_prefetch(Uw + (tid + p * 256) * 16, 0, 1);
    __builtin_prefetch(Ww + (tid + p * 256) * 16, 0, 1);
  }

  // Stage q[:,n,:] (plus transpose) and mem[n] as bf16
  for (int t = tid; t < Bsz * Csz; t += blockDim.x) {
    const int b = t >> 7, c = t & 127;
    const float qv = Q[((size_t)b * Nsz + n) * Csz + c];
    qb[b * QB_LD + c] = (__bf16)qv;
    qT[c * QT_LD + b] = (__bf16)qv;
  }
  for (int t = tid; t < Msz * Csz; t += blockDim.x)
    mb[(t >> 7) * MB_LD + (t & 127)] = (__bf16)Mem[(size_t)n * Msz * Csz + t];
  __syncthreads();

  // scores_u = mem_n (64x128) @ q_n^T (128x256) -> sc (64x256)
  for (int t = w; t < 64; t += 8) {       // 4 x 16 tiles, wave-uniform
    const int m0 = (t & 3) << 4, n0 = (t >> 2) << 4;
    v8f acc = {};
#pragma unroll
    for (int k0 = 0; k0 < Csz; k0 += 32)
      acc = wmma_bf16(frag_a(mb, m0, k0, MB_LD, lane),
                      frag_b(qT, k0, n0, QT_LD, lane), acc);
    store_d(sc, acc, m0, n0, SC_LD, lane);
  }
  __syncthreads();

  // softmax over b (rows of 256) + hard_shrink_relu + l1 -> at (bf16)
  for (int m = w; m < Msz; m += 8) {
    const float* row = sc + m * SC_LD;
    float v[8], mx = -3.4e38f;
#pragma unroll
    for (int j = 0; j < 8; ++j) { v[j] = row[lane + 32 * j]; mx = fmaxf(mx, v[j]); }
    mx = wave_max(mx);
    float s = 0.f;
#pragma unroll
    for (int j = 0; j < 8; ++j) { v[j] = __expf(v[j] - mx); s += v[j]; }
    s = wave_sum(s);
    const float inv = 1.f / s;
    float s2 = 0.f;
#pragma unroll
    for (int j = 0; j < 8; ++j) {
      const float x = v[j] * inv;
      const float d = x - SHRINK;
      const float r = d > 0.f ? d : 0.f;
      v[j] = r * x / (fabsf(d) + EPSV);
      s2 += v[j];
    }
    s2 = wave_sum(s2);
    const float inv2 = 1.f / fmaxf(s2, EPSV);
#pragma unroll
    for (int j = 0; j < 8; ++j)
      at[m * AT_LD + lane + 32 * j] = (__bf16)(v[j] * inv2);
  }
  __syncthreads();

  // stage Uw^T / Ww^T into recycled regions (coalesced global reads)
  for (int t = tid; t < Csz * Csz; t += blockDim.x) {
    const int r = t >> 7, k = t & 127;  // Uw[r][k]
    uwT[k * UW_LD + r] = (__bf16)Uw[t];
    wwT[k * UW_LD + r] = (__bf16)Ww[t];
  }
  __syncthreads();

  // add_mem = attn_u (64x256) @ q_n (256x128) -> amf (f32) + amb (bf16)
  for (int t = w; t < 32; t += 8) {       // 4 x 8 tiles
    const int m0 = (t >> 3) << 4, n0 = (t & 7) << 4;
    v8f acc = {};
#pragma unroll
    for (int k0 = 0; k0 < Bsz; k0 += 32)
      acc = wmma_bf16(frag_a(at, m0, k0, AT_LD, lane),
                      frag_b(qb, k0, n0, QB_LD, lane), acc);
    const int nn = n0 + (lane & 15);
    const int mm = m0 + ((lane >> 4) << 3);
#pragma unroll
    for (int i = 0; i < 8; ++i) {
      amf[(mm + i) * Csz + nn]    = acc[i];
      amb[(mm + i) * AMB_LD + nn] = (__bf16)acc[i];
    }
  }
  __syncthreads();

  // gate = sigmoid(mem@Uw^T + am@Ww^T + Ub + Wb); mem_new = (1-g)*mem + g*am
  for (int t = w; t < 32; t += 8) {
    const int m0 = (t >> 3) << 4, n0 = (t & 7) << 4;
    v8f acc = {};
#pragma unroll
    for (int k0 = 0; k0 < Csz; k0 += 32)
      acc = wmma_bf16(frag_a(mb, m0, k0, MB_LD, lane),
                      frag_b(uwT, k0, n0, UW_LD, lane), acc);
#pragma unroll
    for (int k0 = 0; k0 < Csz; k0 += 32)
      acc = wmma_bf16(frag_a(amb, m0, k0, AMB_LD, lane),
                      frag_b(wwT, k0, n0, UW_LD, lane), acc);
    const int c  = n0 + (lane & 15);
    const int mm = m0 + ((lane >> 4) << 3);
    const float bias = Ub[c] + Wb[c];
#pragma unroll
    for (int i = 0; i < 8; ++i) {
      const int m = mm + i;
      const float g  = 1.f / (1.f + __expf(-(acc[i] + bias)));
      const float mv = Mem[((size_t)n * Msz + m) * Csz + c];  // f32 (L2)
      const float av = amf[m * Csz + c];
      MemNew[((size_t)n * Msz + m) * Csz + c] = (1.f - g) * mv + g * av;
    }
  }
}

// -------- Phase B LDS layout --------
#define MNT_LD 72   // mem_new^T bf16 [128][72]
#define SCB_LD 64   // scores f32 [256][64]
#define ATB_LD 72   // attn bf16 [256][72]
#define B_QB  0
#define B_MN  69632
#define B_MNT 87040
#define B_SC  105472
#define B_AT  171008
#define B_TOTAL 207872

// ---------------- Phase B: read (per n) ----------------
__global__ void mm_read_kernel(const float* __restrict__ Q,
                               const float* __restrict__ MemNew,
                               float* __restrict__ Out,      // (B,N,2C)
                               float* __restrict__ AttnOut)  // (B,N,M)
{
  const int n    = blockIdx.x;
  const int tid  = threadIdx.x;
  const int lane = tid & 31;
  const int w    = tid >> 5;

  __bf16* qb  = (__bf16*)(smemraw + B_QB);
  __bf16* mn  = (__bf16*)(smemraw + B_MN);
  __bf16* mnT = (__bf16*)(smemraw + B_MNT);
  float*  sc  = (float*)(smemraw + B_SC);
  __bf16* at  = (__bf16*)(smemraw + B_AT);

  // stage q (and stream concat-left output), mem_new (+transpose)
  for (int t = tid; t < Bsz * Csz; t += blockDim.x) {
    const int b = t >> 7, c = t & 127;
    const float qv = Q[((size_t)b * Nsz + n) * Csz + c];
    qb[b * QB_LD + c] = (__bf16)qv;
    Out[((size_t)b * Nsz + n) * (2 * Csz) + c] = qv;
  }
  for (int t = tid; t < Msz * Csz; t += blockDim.x) {
    const int m = t >> 7, c = t & 127;
    const float v = MemNew[(size_t)n * Msz * Csz + t];
    mn[m * QB_LD + c]   = (__bf16)v;
    mnT[c * MNT_LD + m] = (__bf16)v;
  }
  __syncthreads();

  // scores_r = q_n (256x128) @ mem_new^T (128x64) -> sc (256x64)
  for (int t = w; t < 64; t += 8) {       // 16 x 4 tiles
    const int m0 = (t >> 2) << 4, n0 = (t & 3) << 4;
    v8f acc = {};
#pragma unroll
    for (int k0 = 0; k0 < Csz; k0 += 32)
      acc = wmma_bf16(frag_a(qb, m0, k0, QB_LD, lane),
                      frag_b(mnT, k0, n0, MNT_LD, lane), acc);
    store_d(sc, acc, m0, n0, SCB_LD, lane);
  }
  __syncthreads();

  // softmax over M (rows of 64) + shrink + l1; emit attn_r output
  for (int r = w; r < Bsz; r += 8) {
    const float* row = sc + r * SCB_LD;
    float v0 = row[lane], v1 = row[lane + 32];
    const float mx = wave_max(fmaxf(v0, v1));
    v0 = __expf(v0 - mx); v1 = __expf(v1 - mx);
    const float s = wave_sum(v0 + v1);
    const float inv = 1.f / s;
    const float x0 = v0 * inv, x1 = v1 * inv;
    const float d0 = x0 - SHRINK, d1 = x1 - SHRINK;
    const float y0 = (d0 > 0.f ? d0 : 0.f) * x0 / (fabsf(d0) + EPSV);
    const float y1 = (d1 > 0.f ? d1 : 0.f) * x1 / (fabsf(d1) + EPSV);
    const float s2 = wave_sum(y0 + y1);
    const float inv2 = 1.f / fmaxf(s2, EPSV);
    const float a0 = y0 * inv2, a1 = y1 * inv2;
    at[r * ATB_LD + lane]      = (__bf16)a0;
    at[r * ATB_LD + lane + 32] = (__bf16)a1;
    float* ao = AttnOut + ((size_t)r * Nsz + n) * Msz;
    ao[lane] = a0; ao[lane + 32] = a1;
  }
  __syncthreads();

  // add_memory = attn_r (256x64) @ mem_new (64x128) -> concat-right of Out
  for (int t = w; t < 128; t += 8) {      // 16 x 8 tiles
    const int m0 = (t >> 3) << 4, n0 = (t & 7) << 4;
    v8f acc = {};
#pragma unroll
    for (int k0 = 0; k0 < Msz; k0 += 32)
      acc = wmma_bf16(frag_a(at, m0, k0, ATB_LD, lane),
                      frag_b(mn, k0, n0, QB_LD, lane), acc);
    const int c  = Csz + n0 + (lane & 15);
    const int bb = m0 + ((lane >> 4) << 3);
#pragma unroll
    for (int i = 0; i < 8; ++i)
      Out[((size_t)(bb + i) * Nsz + n) * (2 * Csz) + c] = acc[i];
  }
}

extern "C" void kernel_launch(void* const* d_in, const int* in_sizes, int n_in,
                              void* d_out, int out_size, void* d_ws, size_t ws_size,
                              hipStream_t stream) {
  (void)in_sizes; (void)n_in; (void)d_ws; (void)ws_size; (void)out_size;
  const float* Q   = (const float*)d_in[0];
  const float* Mem = (const float*)d_in[1];
  const float* Uw  = (const float*)d_in[2];
  const float* Ub  = (const float*)d_in[3];
  const float* Ww  = (const float*)d_in[4];
  const float* Wb  = (const float*)d_in[5];

  float* out0 = (float*)d_out;                        // (B,N,2C)
  float* attn = out0 + (size_t)Bsz * Nsz * 2 * Csz;   // (B,N,M)
  float* mnew = attn + (size_t)Bsz * Nsz * Msz;       // (N,M,C)

  (void)hipFuncSetAttribute((const void*)mm_update_kernel,
                            hipFuncAttributeMaxDynamicSharedMemorySize, A_TOTAL);
  (void)hipFuncSetAttribute((const void*)mm_read_kernel,
                            hipFuncAttributeMaxDynamicSharedMemorySize, B_TOTAL);

  mm_update_kernel<<<Nsz, 256, A_TOTAL, stream>>>(Q, Mem, Uw, Ub, Ww, Wb, mnew);
  mm_read_kernel<<<Nsz, 256, B_TOTAL, stream>>>(Q, mnew, out0, attn);
}
